// Block_72404558676296
// MI455X (gfx1250) — matrix operand order repaired and checked by reading, not compiled, not run on hardware
//
#include <hip/hip_runtime.h>
#include <cstdint>
#include <cstddef>

// ---------------------------------------------------------------------------
// Types for CDNA5 WMMA (wave32): v_wmma_f32_16x16x32_bf16
// ---------------------------------------------------------------------------
typedef __attribute__((ext_vector_type(16))) __bf16 v16bf;
typedef __attribute__((ext_vector_type(8)))  __bf16 v8bf;
typedef __attribute__((ext_vector_type(8)))  float  v8f;

#define DEVFN __device__ __forceinline__

// Problem constants (from reference)
constexpr int CB   = 32;            // batch
constexpr int CN   = 513;           // tokens (G+1)
constexpr int CC   = 384;           // channels
constexpr int CH   = 6;             // heads (hd = 64)
constexpr int CV   = 6;             // views
constexpr int ROWS = CB * CN;       // 16416
constexpr int XTR  = CB * 512;      // 16384 (xt rows)
constexpr int NCLUS  = 2048;
constexpr int NDIM2D = 32 * 14 * 14; // 6272

DEVFN float gelu_f(float x) { return 0.5f * x * (1.0f + erff(x * 0.70710678118654752f)); }

// Monotone order-preserving encoding for float atomicMax on unsigned
DEVFN unsigned enc_f(float f) {
  unsigned b = __float_as_uint(f);
  return (b & 0x80000000u) ? ~b : (b | 0x80000000u);
}
DEVFN float dec_max(unsigned e) {
  if (e == 0u) return 0.0f;                 // untouched segment -> ref maps -inf to 0
  unsigned b = (e & 0x80000000u) ? (e ^ 0x80000000u) : ~e;
  return __uint_as_float(b);
}
DEVFN float seg_val(unsigned e, float s, float cnt) {
  return dec_max(e) + s / fmaxf(cnt, 1.0f);
}

// ---------------------------------------------------------------------------
// Utility kernels
// ---------------------------------------------------------------------------
__global__ void k_fill_f32(float* p, float v, size_t n) {
  size_t i = (size_t)blockIdx.x * 256 + threadIdx.x;
  if (i < n) p[i] = v;
}
__global__ void k_f32_to_bf16(const float* __restrict__ in, __bf16* __restrict__ out, size_t n) {
  size_t i = (size_t)blockIdx.x * 256 + threadIdx.x;
  if (i < n) out[i] = (__bf16)in[i];
}
// transpose-convert: w[K][N] f32  ->  wt[N][K] bf16 (so B-fragment loads are contiguous in K)
__global__ void k_convT(const float* __restrict__ w, __bf16* __restrict__ wt, int K, int N) {
  size_t i = (size_t)blockIdx.x * 256 + threadIdx.x;
  if (i >= (size_t)K * N) return;
  int k = (int)(i / N), n = (int)(i % N);
  wt[(size_t)n * K + k] = (__bf16)w[i];
}

// ---------------------------------------------------------------------------
// Row LayerNorm (C=384), output bf16 (feeds WMMA GEMMs). Block=128, 3 elems/thread.
// ---------------------------------------------------------------------------
__global__ void k_ln(const float* __restrict__ x, const float* __restrict__ g,
                     const float* __restrict__ b, __bf16* __restrict__ out, int rows) {
  int r = blockIdx.x;
  if (r >= rows) return;
  const float* xr = x + (size_t)r * CC;
  float v[3], s = 0.f, s2 = 0.f;
#pragma unroll
  for (int j = 0; j < 3; ++j) {
    int c = threadIdx.x + 128 * j;
    v[j] = xr[c];
    s += v[j]; s2 += v[j] * v[j];
  }
  __shared__ float sh[2][128];
  sh[0][threadIdx.x] = s; sh[1][threadIdx.x] = s2;
  __syncthreads();
  for (int st = 64; st > 0; st >>= 1) {
    if ((int)threadIdx.x < st) {
      sh[0][threadIdx.x] += sh[0][threadIdx.x + st];
      sh[1][threadIdx.x] += sh[1][threadIdx.x + st];
    }
    __syncthreads();
  }
  float mean = sh[0][0] * (1.0f / CC);
  float var  = sh[1][0] * (1.0f / CC) - mean * mean;
  float rstd = rsqrtf(var + 1e-5f);
#pragma unroll
  for (int j = 0; j < 3; ++j) {
    int c = threadIdx.x + 128 * j;
    out[(size_t)r * CC + c] = (__bf16)((v[j] - mean) * rstd * g[c] + b[c]);
  }
}

// ---------------------------------------------------------------------------
// WMMA GEMM, 32x32 tile per wave (2 A-frags x 2 B-frags -> 4 WMMA / k-step).
// out = epi(A[M,K]bf16 @ BT[N,K]bf16):
//   t = acc + bias[col]; if(act) t = gelu(t); t *= scaleVec[col];
//   t *= maskRow[row];   t += residual[row,col]
// Block = 8 waves, each on its own 32x32 tile. M,N must be multiples of 32,
// K multiple of 32 (true for all GEMMs here).
// ---------------------------------------------------------------------------
__global__ void k_gemm_bf16(const __bf16* __restrict__ A, const __bf16* __restrict__ BT,
                            const float* __restrict__ bias, const float* __restrict__ scaleVec,
                            const int* __restrict__ maskRow, const float* __restrict__ residual,
                            float* __restrict__ outF, __bf16* __restrict__ outBf,
                            int M, int N, int K, int actGelu) {
  const int lane = threadIdx.x & 31;
  const int wave = threadIdx.x >> 5;
  const int l16  = lane & 15;
  const int grp  = lane >> 4;
  const int tilesN = N >> 5;                       // 32-wide tiles
  const long total = (long)(M >> 5) * tilesN;
  const long t = (long)blockIdx.x * 8 + wave;
  if (t >= total) return;                          // wave-uniform: EXEC all-ones for WMMA
  const int tm = (int)(t / tilesN);
  const int tn = (int)(t % tilesN);
  const __bf16* aRow0 = A  + (size_t)(tm * 32 + l16) * K;        // A: lane = row M
  const __bf16* aRow1 = aRow0 + (size_t)16 * K;
  const __bf16* bRow0 = BT + (size_t)(tn * 32 + l16) * K;        // B: lane = col N (pre-transposed)
  const __bf16* bRow1 = bRow0 + (size_t)16 * K;
  v8f acc00 = {}, acc01 = {}, acc10 = {}, acc11 = {};
  for (int k0 = 0; k0 < K; k0 += 32) {
    v16bf a0, a1, b0, b1;
    // A 16x32 layout: half-wave 0 holds K {0..7,16..23}, half-wave 1 holds K {8..15,24..31}
    ((v8bf*)&a0)[0] = *(const v8bf*)(aRow0 + k0 + grp * 8);
    ((v8bf*)&a0)[1] = *(const v8bf*)(aRow0 + k0 + 16 + grp * 8);
    ((v8bf*)&a1)[0] = *(const v8bf*)(aRow1 + k0 + grp * 8);
    ((v8bf*)&a1)[1] = *(const v8bf*)(aRow1 + k0 + 16 + grp * 8);
    // B 32x16 layout: half-wave 0 holds K {0..7,16..23}, half-wave 1 holds K {8..15,24..31}
    ((v8bf*)&b0)[0] = *(const v8bf*)(bRow0 + k0 + grp * 8);
    ((v8bf*)&b0)[1] = *(const v8bf*)(bRow0 + k0 + 16 + grp * 8);
    ((v8bf*)&b1)[0] = *(const v8bf*)(bRow1 + k0 + grp * 8);
    ((v8bf*)&b1)[1] = *(const v8bf*)(bRow1 + k0 + 16 + grp * 8);
    acc00 = __builtin_amdgcn_wmma_f32_16x16x32_bf16(false, a0, false, b0, (short)0, acc00, false, false);
    acc01 = __builtin_amdgcn_wmma_f32_16x16x32_bf16(false, a0, false, b1, (short)0, acc01, false, false);
    acc10 = __builtin_amdgcn_wmma_f32_16x16x32_bf16(false, a1, false, b0, (short)0, acc10, false, false);
    acc11 = __builtin_amdgcn_wmma_f32_16x16x32_bf16(false, a1, false, b1, (short)0, acc11, false, false);
  }
  const v8f* accs[4] = { &acc00, &acc01, &acc10, &acc11 };
#pragma unroll
  for (int mi = 0; mi < 2; ++mi) {
#pragma unroll
    for (int nj = 0; nj < 2; ++nj) {
      const v8f& a = *accs[mi * 2 + nj];
      const int col = tn * 32 + nj * 16 + l16;
      const float bv = bias ? bias[col] : 0.0f;
      const float sv = scaleVec ? scaleVec[col] : 1.0f;
#pragma unroll
      for (int r = 0; r < 8; ++r) {
        const int row = tm * 32 + mi * 16 + r + 8 * grp;   // C/D layout: M = vgpr + 8*(lane>=16)
        float v = a[r] + bv;
        if (actGelu) v = gelu_f(v);
        v *= sv;
        if (maskRow) v *= (float)maskRow[row];
        size_t o = (size_t)row * N + col;
        if (residual) v += residual[o];
        if (outF)  outF[o] = v;
        if (outBf) outBf[o] = (__bf16)v;
      }
    }
  }
}

// ---------------------------------------------------------------------------
// Flash attention: one wave per (b, h, 16-row tile). qkv bf16 rows of 1152
// (q|k|v each H*64). Online softmax over keys in steps of 32; QK^T and P@V
// both through v_wmma_f32_16x16x32_bf16. Output written bf16 (proj GEMM input).
// ---------------------------------------------------------------------------
__global__ void k_flash_attn(const __bf16* __restrict__ qkv, __bf16* __restrict__ out) {
  const int lane = threadIdx.x & 31;
  const int l16  = lane & 15;
  const int grp  = lane >> 4;
  int id = blockIdx.x;
  const int mt = id % 33; id /= 33;            // 33 row tiles cover 513 (pad to 528)
  const int h  = id % CH;
  const int b  = id / CH;
  const __bf16* base = qkv + (size_t)b * CN * 1152;
  const int qc = h * 64, kc = 384 + h * 64, vc = 768 + h * 64;

  // Q fragments (A layout), clamp padded rows to a valid address
  int mq = mt * 16 + l16; if (mq > CN - 1) mq = CN - 1;
  const __bf16* qrow = base + (size_t)mq * 1152 + qc;
  v16bf qa[2];
#pragma unroll
  for (int kk = 0; kk < 2; ++kk) {
    ((v8bf*)&qa[kk])[0] = *(const v8bf*)(qrow + kk * 32 + grp * 8);
    ((v8bf*)&qa[kk])[1] = *(const v8bf*)(qrow + kk * 32 + 16 + grp * 8);
  }

  v8f o[4] = {};
  float rmax[8], rsum[8];
#pragma unroll
  for (int r = 0; r < 8; ++r) { rmax[r] = -1e30f; rsum[r] = 0.f; }

  __shared__ __align__(16) __bf16 ptile[16][32];

  for (int n0 = 0; n0 < CN; n0 += 32) {
    // --- scores for 32 keys: 2 key tiles x 2 hd-halves of WMMA ---
    v8f s[2];
#pragma unroll
    for (int t = 0; t < 2; ++t) {
      int kn = n0 + t * 16 + l16; if (kn > CN - 1) kn = CN - 1;
      const __bf16* krow = base + (size_t)kn * 1152 + kc;
      v8f sc = {};
#pragma unroll
      for (int kk = 0; kk < 2; ++kk) {
        v16bf kb;   // B layout: half-wave holds K {8g..8g+7, 16+8g..16+8g+7} (K^T rows are key rows)
        ((v8bf*)&kb)[0] = *(const v8bf*)(krow + kk * 32 + grp * 8);
        ((v8bf*)&kb)[1] = *(const v8bf*)(krow + kk * 32 + 16 + grp * 8);
        sc = __builtin_amdgcn_wmma_f32_16x16x32_bf16(false, qa[kk], false, kb, (short)0, sc, false, false);
      }
      s[t] = sc;
    }
    const bool v0 = (n0 + l16) < CN;
    const bool v1 = (n0 + 16 + l16) < CN;
    float alpha[8];
#pragma unroll
    for (int r = 0; r < 8; ++r) {
      float a = v0 ? s[0][r] * 0.125f : -1e30f;   // hd^-0.5 = 1/8
      float c = v1 ? s[1][r] * 0.125f : -1e30f;
      float ml = fmaxf(a, c);
      for (int m = 8; m >= 1; m >>= 1) ml = fmaxf(ml, __shfl_xor(ml, m, 32));
      float mnew = fmaxf(rmax[r], ml);
      float p0 = __expf(a - mnew);
      float p1 = __expf(c - mnew);
      float ps = p0 + p1;
      for (int m = 8; m >= 1; m >>= 1) ps += __shfl_xor(ps, m, 32);
      float al = __expf(rmax[r] - mnew);
      rsum[r] = rsum[r] * al + ps;
      rmax[r] = mnew;
      alpha[r] = al;
      // stash P (C layout -> LDS tile, rows m = r + 8*grp)
      ptile[r + 8 * grp][l16]      = (__bf16)p0;
      ptile[r + 8 * grp][16 + l16] = (__bf16)p1;
    }
#pragma unroll
    for (int r = 0; r < 8; ++r) { o[0][r] *= alpha[r]; o[1][r] *= alpha[r]; o[2][r] *= alpha[r]; o[3][r] *= alpha[r]; }
    __syncthreads();
    // reload P in A-fragment layout
    v16bf pa;
    {
      const __bf16* pr = &ptile[l16][0];
      ((v8bf*)&pa)[0] = *(const v8bf*)(pr + grp * 8);
      ((v8bf*)&pa)[1] = *(const v8bf*)(pr + 16 + grp * 8);
    }
    __syncthreads();
    // --- P(16x32) @ V(32x64): 4 dim tiles ---
#pragma unroll
    for (int dt = 0; dt < 4; ++dt) {
      v16bf vb;
#pragma unroll
      for (int e = 0; e < 16; ++e) {
        int key = n0 + grp * 8 + (e & 7) + 16 * (e >> 3); if (key > CN - 1) key = CN - 1;  // p==0 for padded keys
        vb[e] = base[(size_t)key * 1152 + vc + dt * 16 + l16];
      }
      o[dt] = __builtin_amdgcn_wmma_f32_16x16x32_bf16(false, pa, false, vb, (short)0, o[dt], false, false);
    }
  }
  // epilogue: divide by row sum, store bf16
#pragma unroll
  for (int r = 0; r < 8; ++r) {
    int m = mt * 16 + r + 8 * grp;
    if (m >= CN) continue;
    float inv = 1.0f / rsum[r];
    size_t orow = ((size_t)b * CN + m) * CC + h * 64;
    out[orow + l16]      = (__bf16)(o[0][r] * inv);
    out[orow + 16 + l16] = (__bf16)(o[1][r] * inv);
    out[orow + 32 + l16] = (__bf16)(o[2][r] * inv);
    out[orow + 48 + l16] = (__bf16)(o[3][r] * inv);
  }
}

// ---------------------------------------------------------------------------
// Fused adapter: out = x1 + xffn + 0.5 * ( gelu(xffn@dw + db) @ uw + ub )
// K=16 bottleneck -> plain VALU. Block=128 per row.
// ---------------------------------------------------------------------------
__global__ void k_adapter(const float* __restrict__ x1, const float* __restrict__ xffn,
                          const float* __restrict__ dw, const float* __restrict__ db,
                          const float* __restrict__ uw, const float* __restrict__ ub,
                          float* __restrict__ out, int rows) {
  int r = blockIdx.x;
  if (r >= rows) return;
  __shared__ float fr[CC];
  __shared__ float mid[16];
  const float* xf = xffn + (size_t)r * CC;
#pragma unroll
  for (int j = 0; j < 3; ++j) fr[threadIdx.x + 128 * j] = xf[threadIdx.x + 128 * j];
  __syncthreads();
  if (threadIdx.x < 16) {
    float a = db[threadIdx.x];
    for (int k = 0; k < CC; ++k) a += fr[k] * dw[k * 16 + threadIdx.x];
    mid[threadIdx.x] = gelu_f(a);
  }
  __syncthreads();
#pragma unroll
  for (int j = 0; j < 3; ++j) {
    int c = threadIdx.x + 128 * j;
    float ad = ub[c];
#pragma unroll
    for (int k = 0; k < 16; ++k) ad += mid[k] * uw[k * CC + c];
    size_t oidx = (size_t)r * CC + c;
    out[oidx] = x1[oidx] + fr[c] + 0.5f * ad;   // SCALE_FACTOR = 0.5
  }
}

// xt rows of full x (skip cls token per batch)
__global__ void k_copy_xt(const float* __restrict__ xfull, float* __restrict__ xt) {
  size_t i = (size_t)blockIdx.x * 256 + threadIdx.x;
  if (i >= (size_t)XTR * CC) return;
  int r = (int)(i / CC), c = (int)(i % CC);
  xt[i] = xfull[((size_t)(r >> 9) * CN + 1 + (r & 511)) * CC + c];
}

// segment max (encoded uint) + sum + count
__global__ void k_scatter(const float* __restrict__ feat, const int* __restrict__ idx,
                          unsigned* __restrict__ smax, float* __restrict__ ssum,
                          float* __restrict__ scnt, int rows) {
  size_t i = (size_t)blockIdx.x * 256 + threadIdx.x;
  if (i >= (size_t)rows * CC) return;
  int r = (int)(i / CC), c = (int)(i % CC);
  int s = idx[r];
  float v = feat[i];
  atomicMax(&smax[(size_t)s * CC + c], enc_f(v));
  atomicAdd(&ssum[(size_t)s * CC + c], v);
  if (c == 0) atomicAdd(&scnt[s], 1.0f);
}

// per-channel batch-norm stats over nseg segment rows of (fixedmax + mean)
__global__ void k_bnstats(const unsigned* __restrict__ smax, const float* __restrict__ ssum,
                          const float* __restrict__ scnt, int nseg,
                          float* __restrict__ meanOut, float* __restrict__ varOut) {
  int c = blockIdx.x;
  float s = 0.f, s2 = 0.f;
  for (int i = threadIdx.x; i < nseg; i += 256) {
    float t = seg_val(smax[(size_t)i * CC + c], ssum[(size_t)i * CC + c], scnt[i]);
    s += t; s2 += t * t;
  }
  __shared__ float sh[2][256];
  sh[0][threadIdx.x] = s; sh[1][threadIdx.x] = s2;
  __syncthreads();
  for (int st = 128; st > 0; st >>= 1) {
    if ((int)threadIdx.x < st) {
      sh[0][threadIdx.x] += sh[0][threadIdx.x + st];
      sh[1][threadIdx.x] += sh[1][threadIdx.x + st];
    }
    __syncthreads();
  }
  if (threadIdx.x == 0) {
    float m = sh[0][0] / nseg;
    meanOut[c] = m;
    varOut[c]  = sh[1][0] / nseg - m * m;
  }
}

// x3d[r,c] = gelu(bn(seg_val))[cluster[r]]
__global__ void k_gather3d(const unsigned* __restrict__ smax, const float* __restrict__ ssum,
                           const float* __restrict__ scnt, const float* __restrict__ mean,
                           const float* __restrict__ var, const float* __restrict__ g,
                           const float* __restrict__ b, const int* __restrict__ cluster,
                           float* __restrict__ x3d) {
  size_t i = (size_t)blockIdx.x * 256 + threadIdx.x;
  if (i >= (size_t)XTR * CC) return;
  int r = (int)(i / CC), c = (int)(i % CC);
  int s = cluster[r];
  float t = seg_val(smax[(size_t)s * CC + c], ssum[(size_t)s * CC + c], scnt[s]);
  x3d[i] = gelu_f((t - mean[c]) * rsqrtf(var[c] + 1e-5f) * g[c] + b[c]);
}

// per view: d2 = gelu(bn(seg_val))[idx]; sim = (cos(d2,x3d)+1)/2;
// acc += sim*d2 ; ssum += sim   (normalize by ssum at the end)
__global__ void k_simacc(const unsigned* __restrict__ smax, const float* __restrict__ ssum_seg,
                         const float* __restrict__ scnt, const float* __restrict__ mean,
                         const float* __restrict__ var, const float* __restrict__ g,
                         const float* __restrict__ b, const int* __restrict__ idx,
                         const float* __restrict__ x3d, float* __restrict__ acc,
                         float* __restrict__ simsum) {
  int r = blockIdx.x;
  if (r >= XTR) return;
  int s = idx[r];
  float d2v[3];
  float dot = 0.f, na = 0.f, nb = 0.f;
#pragma unroll
  for (int j = 0; j < 3; ++j) {
    int c = threadIdx.x + 128 * j;
    float t = seg_val(smax[(size_t)s * CC + c], ssum_seg[(size_t)s * CC + c], scnt[s]);
    float y = gelu_f((t - mean[c]) * rsqrtf(var[c] + 1e-5f) * g[c] + b[c]);
    d2v[j] = y;
    float x3 = x3d[(size_t)r * CC + c];
    dot += y * x3; na += y * y; nb += x3 * x3;
  }
  __shared__ float sh[3][128];
  sh[0][threadIdx.x] = dot; sh[1][threadIdx.x] = na; sh[2][threadIdx.x] = nb;
  __syncthreads();
  for (int st = 64; st > 0; st >>= 1) {
    if ((int)threadIdx.x < st) {
      sh[0][threadIdx.x] += sh[0][threadIdx.x + st];
      sh[1][threadIdx.x] += sh[1][threadIdx.x + st];
      sh[2][threadIdx.x] += sh[2][threadIdx.x + st];
    }
    __syncthreads();
  }
  float cosv = sh[0][0] / (fmaxf(sqrtf(sh[1][0]), 1e-8f) * fmaxf(sqrtf(sh[2][0]), 1e-8f));
  float sim = (cosv + 1.0f) * 0.5f;
#pragma unroll
  for (int j = 0; j < 3; ++j) {
    int c = threadIdx.x + 128 * j;
    acc[(size_t)r * CC + c] += sim * d2v[j];
  }
  if (threadIdx.x == 0) simsum[r] += sim;
}

// d_out(xt rows) += COEF_PRO * acc / simsum
__global__ void k_final_add(const float* __restrict__ acc, const float* __restrict__ simsum,
                            float* __restrict__ out) {
  size_t i = (size_t)blockIdx.x * 256 + threadIdx.x;
  if (i >= (size_t)XTR * CC) return;
  int r = (int)(i / CC), c = (int)(i % CC);
  size_t dr = ((size_t)(r >> 9) * CN + 1 + (r & 511)) * CC + c;
  out[dr] += acc[i] / simsum[r];
}

// ---------------------------------------------------------------------------
// Host orchestration
// ---------------------------------------------------------------------------
static inline int eblocks(size_t n) { return (int)((n + 255) / 256); }

extern "C" void kernel_launch(void* const* d_in, const int* in_sizes, int n_in,
                              void* d_out, int out_size, void* d_ws, size_t ws_size,
                              hipStream_t stream) {
  (void)in_sizes; (void)n_in; (void)out_size; (void)ws_size;
  // inputs in setup_inputs() dict order
  const float* x         = (const float*)d_in[0];
  const int*   cluster   = (const int*)  d_in[1];
  const int*   fgi       = (const int*)  d_in[2];
  const int*   mask      = (const int*)  d_in[3];
  const float* ln1_g     = (const float*)d_in[4];
  const float* ln1_b     = (const float*)d_in[5];
  const float* qkv_w     = (const float*)d_in[6];
  const float* proj_w    = (const float*)d_in[7];
  const float* proj_b    = (const float*)d_in[8];
  const float* ls1_g     = (const float*)d_in[9];
  const float* ln2_g     = (const float*)d_in[10];
  const float* ln2_b     = (const float*)d_in[11];
  const float* fc1_w     = (const float*)d_in[12];
  const float* fc1_b     = (const float*)d_in[13];
  const float* fc2_w     = (const float*)d_in[14];
  const float* fc2_b     = (const float*)d_in[15];
  const float* ls2_g     = (const float*)d_in[16];
  const float* ad_down_w = (const float*)d_in[17];
  const float* ad_down_b = (const float*)d_in[18];
  const float* ad_up_w   = (const float*)d_in[19];
  const float* ad_up_b   = (const float*)d_in[20];
  const float* bn3d_g    = (const float*)d_in[21];
  const float* bn3d_b    = (const float*)d_in[22];
  const float* norm3_g   = (const float*)d_in[23];
  const float* norm3_b   = (const float*)d_in[24];
  const float* attn1_w   = (const float*)d_in[25];
  const float* attn1_b   = (const float*)d_in[26];
  const float* bn2d_g    = (const float*)d_in[27];
  const float* bn2d_b    = (const float*)d_in[28];
  float* out = (float*)d_out;

  // workspace carve-up (256B aligned)
  char* wsb = (char*)d_ws;
  size_t off = 0;
  auto alloc = [&](size_t bytes) -> void* {
    off = (off + 255) & ~(size_t)255;
    void* p = wsb + off;
    off += bytes;
    return p;
  };
  float*  qkv_f   = (float*) alloc((size_t)ROWS * 1152 * 4);
  __bf16* qkv_bf  = (__bf16*)alloc((size_t)ROWS * 1152 * 2);
  __bf16* xn_bf   = (__bf16*)alloc((size_t)ROWS * CC * 2);     // reused for LN1/LN2/norm3
  __bf16* h_bf    = (__bf16*)alloc((size_t)ROWS * 1536 * 2);
  __bf16* attn_bf = (__bf16*)alloc((size_t)ROWS * CC * 2);
  __bf16* wqkvT   = (__bf16*)alloc((size_t)CC * 1152 * 2);
  __bf16* wprojT  = (__bf16*)alloc((size_t)CC * CC * 2);
  __bf16* wfc1T   = (__bf16*)alloc((size_t)CC * 1536 * 2);
  __bf16* wfc2T   = (__bf16*)alloc((size_t)1536 * CC * 2);
  __bf16* wattnT  = (__bf16*)alloc((size_t)CV * CC * CC * 2);
  float*  x1      = (float*) alloc((size_t)ROWS * CC * 4);
  float*  xffn    = (float*) alloc((size_t)ROWS * CC * 4);
  float*  xt      = (float*) alloc((size_t)XTR * CC * 4);
  float*  x3d     = (float*) alloc((size_t)XTR * CC * 4);
  float*  flatx   = (float*) alloc((size_t)XTR * CC * 4);
  float*  accb    = (float*) alloc((size_t)XTR * CC * 4);
  float*  simsum  = (float*) alloc((size_t)XTR * 4);
  unsigned* smax  = (unsigned*)alloc((size_t)NDIM2D * CC * 4);
  float*  ssumseg = (float*) alloc((size_t)NDIM2D * CC * 4);
  float*  scnt    = (float*) alloc((size_t)NDIM2D * 4);
  float*  bnmean  = (float*) alloc(CC * 4);
  float*  bnvar   = (float*) alloc(CC * 4);

  auto fillz = [&](float* p, size_t n) {
    k_fill_f32<<<eblocks(n), 256, 0, stream>>>(p, 0.0f, n);
  };
  auto seg_reset = [&]() {
    fillz((float*)smax, (size_t)NDIM2D * CC);   // bits 0 == empty encoding
    fillz(ssumseg, (size_t)NDIM2D * CC);
    fillz(scnt, NDIM2D);
  };
  auto gemm = [&](const __bf16* A, const __bf16* BT, const float* bias, const float* sv,
                  const int* mrow, const float* res, float* oF, __bf16* oB,
                  int M, int N, int K, int act) {
    long tiles = (long)(M / 32) * (N / 32);
    k_gemm_bf16<<<(int)((tiles + 7) / 8), 256, 0, stream>>>(A, BT, bias, sv, mrow, res, oF, oB, M, N, K, act);
  };

  // --- weight down-convert + transpose (bf16, N-major) ---
  k_convT<<<eblocks((size_t)CC * 1152), 256, 0, stream>>>(qkv_w, wqkvT, CC, 1152);
  k_convT<<<eblocks((size_t)CC * CC),   256, 0, stream>>>(proj_w, wprojT, CC, CC);
  k_convT<<<eblocks((size_t)CC * 1536), 256, 0, stream>>>(fc1_w, wfc1T, CC, 1536);
  k_convT<<<eblocks((size_t)1536 * CC), 256, 0, stream>>>(fc2_w, wfc2T, 1536, CC);
  for (int v = 0; v < CV; ++v)
    k_convT<<<eblocks((size_t)CC * CC), 256, 0, stream>>>(attn1_w + (size_t)v * CC * CC,
                                                          wattnT + (size_t)v * CC * CC, CC, CC);

  // --- transformer block ---
  k_ln<<<ROWS, 128, 0, stream>>>(x, ln1_g, ln1_b, xn_bf, ROWS);
  gemm(xn_bf, wqkvT, nullptr, nullptr, nullptr, nullptr, qkv_f, nullptr, ROWS, 1152, CC, 0);
  k_f32_to_bf16<<<eblocks((size_t)ROWS * 1152), 256, 0, stream>>>(qkv_f, qkv_bf, (size_t)ROWS * 1152);
  k_flash_attn<<<CB * CH * 33, 32, 0, stream>>>(qkv_bf, attn_bf);
  gemm(attn_bf, wprojT, proj_b, ls1_g, nullptr, x, x1, nullptr, ROWS, CC, CC, 0);        // x1 = x + ls1*(attn@W+b)
  k_ln<<<ROWS, 128, 0, stream>>>(x1, ln2_g, ln2_b, xn_bf, ROWS);
  gemm(xn_bf, wfc1T, fc1_b, nullptr, nullptr, nullptr, nullptr, h_bf, ROWS, 1536, CC, 1); // gelu, bf16 out
  gemm(h_bf, wfc2T, fc2_b, ls2_g, nullptr, nullptr, xffn, nullptr, ROWS, CC, 1536, 0);    // x_ffn
  k_adapter<<<ROWS, 128, 0, stream>>>(x1, xffn, ad_down_w, ad_down_b, ad_up_w, ad_up_b, out, ROWS);

  // --- cluster (3D) branch ---
  k_copy_xt<<<eblocks((size_t)XTR * CC), 256, 0, stream>>>(out, xt);
  seg_reset();
  k_scatter<<<eblocks((size_t)XTR * CC), 256, 0, stream>>>(xt, cluster, smax, ssumseg, scnt, XTR);
  k_bnstats<<<CC, 256, 0, stream>>>(smax, ssumseg, scnt, NCLUS, bnmean, bnvar);
  k_gather3d<<<eblocks((size_t)XTR * CC), 256, 0, stream>>>(smax, ssumseg, scnt, bnmean, bnvar,
                                                            bn3d_g, bn3d_b, cluster, x3d);

  // --- per-view branches, fused cosine weighting ---
  fillz(accb, (size_t)XTR * CC);
  fillz(simsum, XTR);
  for (int v = 0; v < CV; ++v) {
    const int* fgi_v  = fgi  + (size_t)v * XTR;
    const int* mask_v = mask + (size_t)v * XTR;
    k_ln<<<XTR, 128, 0, stream>>>(xt, norm3_g + (size_t)v * CC, norm3_b + (size_t)v * CC, xn_bf, XTR);
    gemm(xn_bf, wattnT + (size_t)v * CC * CC, attn1_b + (size_t)v * CC,
         nullptr, mask_v, xt, flatx, nullptr, XTR, CC, CC, 0);        // flat_x = mask*(h)+xt
    seg_reset();
    k_scatter<<<eblocks((size_t)XTR * CC), 256, 0, stream>>>(flatx, fgi_v, smax, ssumseg, scnt, XTR);
    k_bnstats<<<CC, 256, 0, stream>>>(smax, ssumseg, scnt, NDIM2D, bnmean, bnvar);
    k_simacc<<<XTR, 128, 0, stream>>>(smax, ssumseg, scnt, bnmean, bnvar,
                                      bn2d_g + (size_t)v * CC, bn2d_b + (size_t)v * CC,
                                      fgi_v, x3d, accb, simsum);
  }
  k_final_add<<<eblocks((size_t)XTR * CC), 256, 0, stream>>>(accb, simsum, out);
}
